// MoELayer_51745765982346
// MI455X (gfx1250) — matrix operand order repaired and checked by reading, not compiled, not run on hardware
//
#include <hip/hip_runtime.h>
#include <hip/hip_bf16.h>

#define T_TOKENS 16384
#define DM 1024
#define DF 4096
#define NE 8

#define TILE_M 64
#define TILES_PER_E (T_TOKENS / TILE_M)  // 256
#define FC 64    // d_ff chunk per fused step
#define KP 128   // k-panel staged per phase-1 step (8 panels per chunk)

// LDS row strides (in ushorts), padded for conflict-free ds_load_b128
#define XS  1032   // X tile rows (1024 + 8)
#define W2S 72     // W2 chunk rows (64 + 8)
#define W1S 136    // W1 panel rows (128 + 8)
#define HS  72     // H rows (64 + 8)

// LDS byte offsets (dynamic-shared assumed at LDS offset 0 for TDM descriptors)
#define OFF_X    0
#define OFF_W2   (64 * XS * 2)                    // 132096
#define OFF_W1A  (OFF_W2 + 1024 * W2S * 2)        // 279552
#define OFF_W1B  (OFF_W1A + 64 * W1S * 2)         // 296960
#define OFF_H    (OFF_W1B + 64 * W1S * 2)         // 314368
#define OFF_WT   (OFF_H + 64 * HS * 2)            // 323584
#define OFF_TOK  (OFF_WT + 64 * 4)
#define SMEM_BYTES (OFF_TOK + 64 * 4)             // 324096 <= 320KB WGP budget

typedef __attribute__((ext_vector_type(16))) __bf16 v16bf;
typedef __attribute__((ext_vector_type(8)))  float  v8f;
typedef __attribute__((ext_vector_type(8)))  unsigned short ushort8v;
typedef __attribute__((ext_vector_type(4)))  unsigned short ushort4v;
typedef __attribute__((ext_vector_type(4)))  unsigned int v4u;
typedef __attribute__((ext_vector_type(8)))  int v8i;
typedef __attribute__((ext_vector_type(4)))  int v4i;

#if __has_builtin(__builtin_amdgcn_tensor_load_to_lds) && \
    __has_builtin(__builtin_amdgcn_s_wait_tensorcnt)
#define USE_TDM 1
#else
#define USE_TDM 0
#endif

union BFrag {
    ushort8v h[2];
    v16bf    v;
};

__device__ __forceinline__ unsigned short f2bf(float f) {
    unsigned int u = __float_as_uint(f);
    u += 0x7FFFu + ((u >> 16) & 1u);   // round-to-nearest-even
    return (unsigned short)(u >> 16);
}

#if USE_TDM
// Issue one TDM load: contiguous bf16 block -> LDS with hardware padding.
// tile/tensor dims in 2-byte elements; pad codes per D# spec.
__device__ __forceinline__ void tdm_load(unsigned int lds_off, const void* gaddr,
                                         unsigned int tile_d0, unsigned int tile_d1,
                                         unsigned int stride0,
                                         unsigned int pad_int_code,
                                         unsigned int pad_amt_code) {
    unsigned long long ga = (unsigned long long)(size_t)gaddr;
    v4u g0;
    g0[0] = 1u;                                        // count=1 (valid user D#)
    g0[1] = lds_off;                                   // lds_addr
    g0[2] = (unsigned int)(ga & 0xFFFFFFFFu);          // global_addr lo
    g0[3] = (unsigned int)((ga >> 32) & 0x1FFFFFFu) | (2u << 30);  // addr hi | type=2
    unsigned int w0 = (1u << 16)                       // data_size = 2 bytes
                    | (1u << 20)                       // pad_enable
                    | (pad_int_code << 22)
                    | (pad_amt_code << 25);
    v8i g1;
    g1[0] = (int)w0;
    g1[1] = (int)((tile_d0 & 0xFFFFu) << 16);          // tensor_dim0 = tile_d0 (lo16)
    g1[2] = (int)(((tile_d0 >> 16) & 0xFFFFu) | ((tile_d1 & 0xFFFFu) << 16)); // dim0 hi | dim1 lo
    g1[3] = (int)((tile_d0 & 0xFFFFu) << 16);          // dim1 hi=0 | tile_dim0
    g1[4] = (int)(tile_d1 & 0xFFFFu);                  // tile_dim1 | tile_dim2=0
    g1[5] = (int)stride0;                              // tensor_dim0_stride lo
    g1[6] = 0;
    g1[7] = 0;
    v4i z4 = {0, 0, 0, 0};
    v8i z8 = {0, 0, 0, 0, 0, 0, 0, 0};
    __builtin_amdgcn_tensor_load_to_lds(g0, g1, z4, z4, z8, 0);
}
#endif

// ---------------------------------------------------------------------------
// Router: one wave per token. logits = x@Wr + br -> softmax -> top-1.
// ---------------------------------------------------------------------------
__global__ void router_kernel(const float* __restrict__ x,
                              const float* __restrict__ Wr,
                              const float* __restrict__ br,
                              float* __restrict__ wgt,
                              int* __restrict__ eid) {
    int gid  = blockIdx.x * blockDim.x + threadIdx.x;
    int t    = gid >> 5;
    int lane = gid & 31;
    if (t >= T_TOKENS) return;
    const float* xr = x + (size_t)t * DM;
    float acc[NE];
#pragma unroll
    for (int e = 0; e < NE; ++e) acc[e] = 0.f;
    for (int k = lane; k < DM; k += 32) {
        float xv = xr[k];
        const float* w = Wr + (size_t)k * NE;
#pragma unroll
        for (int e = 0; e < NE; ++e) acc[e] += xv * w[e];
    }
#pragma unroll
    for (int off = 16; off > 0; off >>= 1) {
#pragma unroll
        for (int e = 0; e < NE; ++e) acc[e] += __shfl_xor(acc[e], off, 32);
    }
    if (lane == 0) {
        float m = -1e30f; int am = 0;
#pragma unroll
        for (int e = 0; e < NE; ++e) {
            acc[e] += br[e];
            if (acc[e] > m) { m = acc[e]; am = e; }
        }
        float s = 0.f;
#pragma unroll
        for (int e = 0; e < NE; ++e) s += __expf(acc[e] - m);
        wgt[t] = 1.0f / s;
        eid[t] = am;
    }
}

// ---------------------------------------------------------------------------
// Bucketing.
// ---------------------------------------------------------------------------
__global__ void zero_kernel(int* __restrict__ counts) {
    if (threadIdx.x < NE) counts[threadIdx.x] = 0;
}
__global__ void hist_kernel(const int* __restrict__ eid, int* __restrict__ counts) {
    int t = blockIdx.x * blockDim.x + threadIdx.x;
    if (t < T_TOKENS) atomicAdd(&counts[eid[t]], 1);
}
__global__ void scan_kernel(const int* __restrict__ counts,
                            int* __restrict__ offs, int* __restrict__ cursor) {
    if (threadIdx.x == 0 && blockIdx.x == 0) {
        int s = 0;
        for (int e = 0; e < NE; ++e) { offs[e] = s; cursor[e] = s; s += counts[e]; }
    }
}
__global__ void scatter_kernel(const int* __restrict__ eid,
                               int* __restrict__ cursor, int* __restrict__ perm) {
    int t = blockIdx.x * blockDim.x + threadIdx.x;
    if (t < T_TOKENS) {
        int p = atomicAdd(&cursor[eid[t]], 1);
        perm[p] = t;
    }
}

// ---------------------------------------------------------------------------
// Weight pre-conversion fp32 -> bf16, blocked + transposed so the FFN kernel
// streams contiguous panels (TDM-friendly).
// W1t block (e, cb in 0..63, kbi in 0..7): [64 cols][128 k], k fastest.
// ---------------------------------------------------------------------------
__global__ void w1cvt_kernel(const float* __restrict__ W1, unsigned short* __restrict__ W1t) {
    __shared__ unsigned short sT[64 * W1S];
    const int bx  = blockIdx.x;            // 8*64*8 = 4096 blocks
    const int e   = bx >> 9;
    const int cb  = (bx >> 3) & 63;
    const int kbi = bx & 7;
    const int tid = threadIdx.x;           // 256 threads
#pragma unroll 4
    for (int i = 0; i < 32; ++i) {
        int idx = tid + i * 256;           // 0..8191
        int k = idx >> 6, c = idx & 63;
        float v = W1[((size_t)e * DM + kbi * 128 + k) * DF + cb * 64 + c];
        sT[c * W1S + k] = f2bf(v);
    }
    __syncthreads();
    unsigned short* dst = W1t + ((size_t)((e * 64 + cb) * 8 + kbi)) * 8192;
#pragma unroll 4
    for (int i = 0; i < 32; ++i) {
        int idx = tid + i * 256;
        int c = idx >> 7, k = idx & 127;
        dst[idx] = sT[c * W1S + k];
    }
}

// W2t block (e, c in 0..63): [1024 cols][64 k], k fastest. Grid splits cols by 256.
__global__ void w2cvt_kernel(const float* __restrict__ W2, unsigned short* __restrict__ W2t) {
    __shared__ unsigned short sT[256 * W2S];
    const int bx  = blockIdx.x;            // 8*64*4 = 2048 blocks
    const int e   = bx >> 8;
    const int c   = (bx >> 2) & 63;
    const int cb2 = bx & 3;
    const int tid = threadIdx.x;           // 256 threads
#pragma unroll 4
    for (int i = 0; i < 64; ++i) {
        int idx = tid + i * 256;           // 0..16383
        int kk = idx >> 8, cc = idx & 255;
        float v = W2[((size_t)e * DF + c * 64 + kk) * DM + cb2 * 256 + cc];
        sT[cc * W2S + kk] = f2bf(v);
    }
    __syncthreads();
    unsigned short* dst = W2t + ((size_t)((e * 64 + c) * 1024 + cb2 * 256)) * 64;
#pragma unroll 4
    for (int i = 0; i < 64; ++i) {
        int idx = tid + i * 256;
        int cc = idx >> 6, kk = idx & 63;
        dst[idx] = sT[cc * W2S + kk];
    }
}

// ---------------------------------------------------------------------------
// Fused expert FFN, bf16 WMMA, TDM-staged weights.
// __launch_bounds__(512, 1): one resident WG (forced anyway by 324KB LDS) so
// the allocator gives each wave the full 4-waves/SIMD VGPR budget (no spills).
// ---------------------------------------------------------------------------
__global__ void __launch_bounds__(512, 1)
moe_ffn_kernel(const float* __restrict__ x,
               const unsigned short* __restrict__ W1t,
               const float* __restrict__ b1,
               const unsigned short* __restrict__ W2t,
               const float* __restrict__ b2,
               const float* __restrict__ wgt,
               const int* __restrict__ counts,
               const int* __restrict__ offs,
               const int* __restrict__ perm,
               float* __restrict__ out) {
    const int e    = blockIdx.x >> 8;
    const int tile = blockIdx.x & (TILES_PER_E - 1);
    const int cnt  = counts[e];
    const int row0 = tile * TILE_M;
    if (row0 >= cnt) return;
    const int base = offs[e];

    extern __shared__ unsigned char smem_raw[];
    unsigned short* sX  = (unsigned short*)(smem_raw + OFF_X);
    unsigned short* sW2 = (unsigned short*)(smem_raw + OFF_W2);
    unsigned short* sW1 = (unsigned short*)(smem_raw + OFF_W1A);   // double buffer
    unsigned short* sH  = (unsigned short*)(smem_raw + OFF_H);
    float* sWt = (float*)(smem_raw + OFF_WT);
    int*   sTok = (int*)(smem_raw + OFF_TOK);

    const int tid  = threadIdx.x;
    const int lane = tid & 31;
    const int wid  = tid >> 5;          // 0..15
    const int half = (lane >> 4) & 1;
    const int ln   = lane & 15;

    const unsigned short* W1e = W1t + (size_t)e * 64 * 8 * 8192;   // blocked
    const unsigned short* W2e = W2t + (size_t)e * 64 * 1024 * 64;  // blocked

    if (tid < TILE_M) {
        int r  = row0 + tid;
        int rc = r < cnt ? r : (cnt - 1);
        int tok = perm[base + rc];
        sTok[tid] = tok;
        sWt[tid]  = wgt[tok];
    }
    __syncthreads();

    // Stage gathered X tile fp32 -> bf16
    {
        int r  = tid >> 3;
        int cg = (tid & 7) * 128;
        const float* xr = x + (size_t)sTok[r] * DM;
#pragma unroll 4
        for (int j = 0; j < 32; ++j) {
            int col = cg + j * 4;
            float4 v = *(const float4*)(xr + col);
            ushort4v h;
            h[0] = f2bf(v.x); h[1] = f2bf(v.y); h[2] = f2bf(v.z); h[3] = f2bf(v.w);
            *(ushort4v*)(sX + r * XS + col) = h;
        }
    }

    v8f accY[4][4];
#pragma unroll
    for (int i = 0; i < 4; ++i)
#pragma unroll
        for (int j = 0; j < 4; ++j)
#pragma unroll
            for (int k = 0; k < 8; ++k) accY[i][j][k] = 0.f;

    const int mt1 = wid >> 2, nt1 = wid & 3;

    for (int cc0 = 0; cc0 < 64; ++cc0) {         // d_ff chunk index (FC=64 wide)
        const unsigned short* w1chunk = W1e + (size_t)(cc0 * 8) * 8192; // 8 k-panels
        const unsigned short* w2chunk = W2e + (size_t)cc0 * 1024 * 64;

#if USE_TDM
        if (wid == 0 && lane == 0) {
            // W2 chunk: 8 rows x 8192 elems, pad 4DW per 32DW.
            tdm_load(OFF_W2, w2chunk, 8192u, 8u, 8192u, 4u, 3u);
        }
        if (wid == 1 && lane == 0) {
            // Prologue: panel 0 -> buffer 0. 8192 elems, pad 4DW per 64DW.
            tdm_load(OFF_W1A, w1chunk, 8192u, 0u, 8192u, 5u, 3u);
        }
#endif
        v8f cH;
#pragma unroll
        for (int k = 0; k < 8; ++k) cH[k] = 0.f;

#pragma unroll
        for (int kb = 0; kb < 8; ++kb) {
            const int buf = kb & 1;
#if USE_TDM
            if (wid == 1) {
                if (kb < 7) {
                    if (lane == 0)
                        tdm_load((kb & 1) ? OFF_W1A : OFF_W1B,   // next panel -> other buf
                                 w1chunk + (size_t)(kb + 1) * 8192,
                                 8192u, 0u, 8192u, 5u, 3u);
                    __builtin_amdgcn_s_wait_tensorcnt(1);
                } else {
                    __builtin_amdgcn_s_wait_tensorcnt(0);
                }
            }
            __syncthreads();    // panel kb ready in sW1[buf]
#else
            __syncthreads();    // previous readers of sW1 done
            {   // manual copy: 8192 ushorts, 16 per thread
                int c = tid >> 3, k0 = (tid & 7) * 16;
                const ushort8v* src = (const ushort8v*)(w1chunk + (size_t)kb * 8192 + c * 128 + k0);
                ushort8v* dst = (ushort8v*)(sW1 + c * W1S + k0);
                dst[0] = src[0];
                dst[1] = src[1];
            }
            __syncthreads();
#endif
            const unsigned short* sW1b = sW1 + (USE_TDM ? buf : 0) * (64 * W1S);
#pragma unroll
            for (int kk = 0; kk < KP; kk += 32) {
                BFrag a, b;
                const unsigned short* ap = sX + (mt1 * 16 + ln) * XS + (kb * KP + kk) + half * 8;
                a.h[0] = *(const ushort8v*)ap;
                a.h[1] = *(const ushort8v*)(ap + 16);
                const unsigned short* bp = sW1b + (nt1 * 16 + ln) * W1S + kk + half * 8;
                b.h[0] = *(const ushort8v*)bp;
                b.h[1] = *(const ushort8v*)(bp + 16);
                cH = __builtin_amdgcn_wmma_f32_16x16x32_bf16(
                        false, a.v, false, b.v, (short)0, cH, false, false);
            }
            __syncthreads();    // compute done; next panel may overwrite other buffer
        }

        {   // bias + relu + store H as bf16
            float bias = b1[(size_t)e * DF + cc0 * FC + nt1 * 16 + ln];
            unsigned short* hp = sH + (mt1 * 16 + half * 8) * HS + nt1 * 16 + ln;
#pragma unroll
            for (int v = 0; v < 8; ++v) {
                float hv = cH[v] + bias;
                hv = hv > 0.f ? hv : 0.f;
                hp[v * HS] = f2bf(hv);
            }
        }
        __syncthreads();

#if USE_TDM
        if (wid == 0) __builtin_amdgcn_s_wait_tensorcnt(0);   // W2 chunk landed
        __syncthreads();
#else
        {   // manual copy: 65536 ushorts, 128 per thread
#pragma unroll
            for (int j = 0; j < 8; ++j) {
                int idx = tid + j * 512;
                int c2 = idx >> 2, k0 = (idx & 3) * 16;
                const ushort8v* src = (const ushort8v*)(w2chunk + (size_t)c2 * 64 + k0);
                ushort8v* dst = (ushort8v*)(sW2 + c2 * W2S + k0);
                dst[0] = src[0];
                dst[1] = src[1];
            }
        }
        __syncthreads();
#endif

        // -------- phase 2: accY += H @ W2chunk --------
        // Preload all 4 A fragments per k-step (reused by all col tiles),
        // keep only ONE B fragment live at a time -> low register pressure.
#pragma unroll
        for (int kk = 0; kk < FC; kk += 32) {
            BFrag afr[4];
#pragma unroll
            for (int mt = 0; mt < 4; ++mt) {
                const unsigned short* ap = sH + (mt * 16 + ln) * HS + kk + half * 8;
                afr[mt].h[0] = *(const ushort8v*)ap;
                afr[mt].h[1] = *(const ushort8v*)(ap + 16);
            }
#pragma unroll
            for (int nt = 0; nt < 4; ++nt) {
                BFrag b;
                const unsigned short* bp =
                    sW2 + (wid * 64 + nt * 16 + ln) * W2S + kk + half * 8;
                b.h[0] = *(const ushort8v*)bp;
                b.h[1] = *(const ushort8v*)(bp + 16);
#pragma unroll
                for (int mt = 0; mt < 4; ++mt)
                    accY[mt][nt] = __builtin_amdgcn_wmma_f32_16x16x32_bf16(
                        false, afr[mt].v, false, b.v, (short)0, accY[mt][nt], false, false);
            }
        }
        __syncthreads();    // chunk done: sH / sW2 / sW1 free for next chunk
    }

    // Epilogue: add b2, scale by router weight, scatter rows to out.
#pragma unroll
    for (int nt = 0; nt < 4; ++nt) {
        int col = wid * 64 + nt * 16 + ln;
        float bv = b2[(size_t)e * DM + col];
#pragma unroll
        for (int mt = 0; mt < 4; ++mt) {
#pragma unroll
            for (int v = 0; v < 8; ++v) {
                int row = mt * 16 + half * 8 + v;
                if (row0 + row < cnt) {
                    float y = (accY[mt][nt][v] + bv) * sWt[row];
                    out[(size_t)sTok[row] * DM + col] = y;
                }
            }
        }
    }
}

// ---------------------------------------------------------------------------
extern "C" void kernel_launch(void* const* d_in, const int* in_sizes, int n_in,
                              void* d_out, int out_size, void* d_ws, size_t ws_size,
                              hipStream_t stream) {
    (void)in_sizes; (void)n_in; (void)out_size; (void)ws_size;
    const float* x  = (const float*)d_in[0];
    const float* Wr = (const float*)d_in[1];
    const float* br = (const float*)d_in[2];
    const float* W1 = (const float*)d_in[3];
    const float* b1 = (const float*)d_in[4];
    const float* W2 = (const float*)d_in[5];
    const float* b2 = (const float*)d_in[6];
    float* out = (float*)d_out;

    unsigned char* ws = (unsigned char*)d_ws;
    float* wgt  = (float*)ws;                       // T floats
    int* eid    = (int*)(wgt + T_TOKENS);           // T ints
    int* counts = eid + T_TOKENS;                   // NE
    int* offs   = counts + NE;
    int* cursor = offs + NE;
    int* perm   = cursor + NE;                      // T ints
    unsigned short* W1t = (unsigned short*)(ws + (256u * 1024u));          // 64 MB
    unsigned short* W2t = (unsigned short*)(ws + (256u * 1024u) + (64u * 1024u * 1024u)); // 64 MB

    // Weight pre-convert (blocked bf16) — overlaps router work on the queue.
    w1cvt_kernel<<<NE * 64 * 8, 256, 0, stream>>>(W1, W1t);
    w2cvt_kernel<<<NE * 64 * 4, 256, 0, stream>>>(W2, W2t);

    router_kernel<<<(T_TOKENS * 32) / 256, 256, 0, stream>>>(x, Wr, br, wgt, eid);
    zero_kernel<<<1, 32, 0, stream>>>(counts);
    hist_kernel<<<T_TOKENS / 256, 256, 0, stream>>>(eid, counts);
    scan_kernel<<<1, 1, 0, stream>>>(counts, offs, cursor);
    scatter_kernel<<<T_TOKENS / 256, 256, 0, stream>>>(eid, cursor, perm);

    moe_ffn_kernel<<<NE * TILES_PER_E, 512, SMEM_BYTES, stream>>>(
        x, W1t, b1, W2t, b2, wgt, counts, offs, perm, out);
}
